// MoERouterLayer_73134703117020
// MI455X (gfx1250) — matrix operand order repaired and checked by reading, not compiled, not run on hardware
//
#include <hip/hip_runtime.h>
#include <math.h>

#define T_TOK 2048
#define DH    512
#define NEXP  8
#define INNER 2048
#define F1    4096   // 2*INNER

typedef __attribute__((ext_vector_type(16))) __bf16 v16bf;
typedef __attribute__((ext_vector_type(8)))  float  v8f;

struct Bits256 { uint4 lo, hi; };

// Build a 16x bf16 WMMA fragment from two 16B chunks at p and p+second_off (elems).
__device__ __forceinline__ v16bf load_frag(const __bf16* p, int second_off) {
    Bits256 r;
    r.lo = *(const uint4*)(p);
    r.hi = *(const uint4*)(p + second_off);
    return __builtin_bit_cast(v16bf, r);
}

// ---------------------------------------------------------------- zero y + counters
__global__ void moe_zero_kernel(float* __restrict__ y, int* __restrict__ cnt, int n) {
    int i = blockIdx.x * blockDim.x + threadIdx.x;
    if (blockIdx.x == 0 && threadIdx.x < NEXP) cnt[threadIdx.x] = 0;
    if (i < n) y[i] = 0.0f;
}

// ---------------------------------------------------------- fp32 [R,C] -> bf16 [C,R]
__global__ __launch_bounds__(256)
void transpose_cvt_kernel(const float* __restrict__ in, __bf16* __restrict__ out,
                          int R, int C) {
    __shared__ float tile[32][33];
    const size_t eoff = (size_t)blockIdx.z * (size_t)R * (size_t)C;
    in  += eoff;
    out += eoff;
    const int c0 = blockIdx.x * 32;
    const int r0 = blockIdx.y * 32;
    for (int j = threadIdx.y; j < 32; j += 8)
        tile[j][threadIdx.x] = in[(size_t)(r0 + j) * C + c0 + threadIdx.x];
    __syncthreads();
    for (int j = threadIdx.y; j < 32; j += 8)
        out[(size_t)(c0 + j) * R + r0 + threadIdx.x] = (__bf16)tile[threadIdx.x][j];
}

// ------------------------------------------------------------------------- router
// One wave32 per token: logits, softmax, top-2, append to per-expert lists.
__global__ __launch_bounds__(256)
void moe_router_kernel(const float* __restrict__ x, const float* __restrict__ Wc,
                       const float* __restrict__ bc, float* __restrict__ logits,
                       int* __restrict__ cnt, int* __restrict__ tokl,
                       float* __restrict__ wgtl) {
    const int lane = threadIdx.x & 31;
    const int t = blockIdx.x * 8 + (threadIdx.x >> 5);
    float acc[NEXP];
#pragma unroll
    for (int e = 0; e < NEXP; ++e) acc[e] = 0.0f;
    const float* xr = x + (size_t)t * DH;
    for (int d = lane; d < DH; d += 32) {
        float xv = xr[d];
        const float4* wr = (const float4*)(Wc + (size_t)d * NEXP);
        float4 w0 = wr[0], w1 = wr[1];
        acc[0] += xv * w0.x; acc[1] += xv * w0.y;
        acc[2] += xv * w0.z; acc[3] += xv * w0.w;
        acc[4] += xv * w1.x; acc[5] += xv * w1.y;
        acc[6] += xv * w1.z; acc[7] += xv * w1.w;
    }
#pragma unroll
    for (int off = 16; off > 0; off >>= 1)
#pragma unroll
        for (int e = 0; e < NEXP; ++e)
            acc[e] += __shfl_down(acc[e], off, 32);
    if (lane == 0) {
        float l[NEXP], p[NEXP];
        float mx = -3.0e38f;
#pragma unroll
        for (int e = 0; e < NEXP; ++e) {
            l[e] = acc[e] + bc[e];
            logits[(size_t)t * NEXP + e] = l[e];
            mx = fmaxf(mx, l[e]);
        }
        float s = 0.0f;
#pragma unroll
        for (int e = 0; e < NEXP; ++e) { p[e] = __expf(l[e] - mx); s += p[e]; }
        float inv = 1.0f / s;
#pragma unroll
        for (int e = 0; e < NEXP; ++e) p[e] *= inv;
        int i0 = 0;
#pragma unroll
        for (int e = 1; e < NEXP; ++e) if (p[e] > p[i0]) i0 = e;
        int i1 = (i0 == 0) ? 1 : 0;
#pragma unroll
        for (int e = 0; e < NEXP; ++e) if (e != i0 && p[e] > p[i1]) i1 = e;
        int s0 = atomicAdd(&cnt[i0], 1);
        tokl[i0 * T_TOK + s0] = t; wgtl[i0 * T_TOK + s0] = p[i0];
        int s1 = atomicAdd(&cnt[i1], 1);
        tokl[i1 * T_TOK + s1] = t; wgtl[i1 * T_TOK + s1] = p[i1];
    }
}

// ------------------------------------------------- grouped expert FFN (bf16 WMMA)
// Block = 256 threads (8 waves). Tile = 16 gathered tokens of expert blockIdx.y.
__global__ __launch_bounds__(256)
void moe_ffn_kernel(const float* __restrict__ x,
                    const __bf16* __restrict__ W1t,   // [E][F1][DH] bf16 (K contiguous)
                    const float*  __restrict__ b1,    // [E][F1]
                    const __bf16* __restrict__ W2t,   // [E][DH][INNER] bf16
                    const float*  __restrict__ b2,    // [E][DH]
                    const int*    __restrict__ cnt,
                    const int*    __restrict__ tokl,
                    const float*  __restrict__ wgtl,
                    float* __restrict__ y) {
    __shared__ __align__(16) __bf16 xs[16 * 520];   // 16 x 512 (+8 pad)
    __shared__ __align__(16) __bf16 gs[16 * 2056];  // 16 x 2048 (+8 pad)
    __shared__ int   stok[16];
    __shared__ float swt[16];

    const int e    = blockIdx.y;
    const int tile = blockIdx.x;
    const int n_e  = cnt[e];
    if (tile * 16 >= n_e) return;   // uniform early-out

    const int tid  = threadIdx.x;
    const int lane = tid & 31;
    const int w    = tid >> 5;
    const int half = lane >> 4;     // 0: lanes 0-15, 1: lanes 16-31
    const int l15  = lane & 15;

    if (tid < 16) {
        int slot = tile * 16 + tid;
        if (slot < n_e) {
            stok[tid] = tokl[e * T_TOK + slot];
            swt[tid]  = wgtl[e * T_TOK + slot];
        } else {
            stok[tid] = 0;
            swt[tid]  = 0.0f;
        }
    }
    __syncthreads();

    // Stage gathered x rows as bf16 into LDS.
    for (int i = tid; i < 16 * DH; i += 256) {
        int r = i >> 9, c = i & (DH - 1);
        xs[r * 520 + c] = (__bf16)x[(size_t)stok[r] * DH + c];
    }
    __syncthreads();

    const __bf16* W1e = W1t + (size_t)e * F1 * DH;
    const float*  b1e = b1 + (size_t)e * F1;

    // GEMM1: h = x @ W1  (16 x 512 x 4096), fused bias + SiLU-GLU -> gs (bf16)
    for (int nt = w; nt < INNER / 16; nt += 8) {
        const int na = nt * 16;          // 'a' column tile
        const int ng = na + INNER;       // matching 'gate' column tile
        v8f acc_a = {}; v8f acc_g = {};
        const __bf16* ap  = xs + l15 * 520 + half * 16;
        const __bf16* bpa = W1e + (size_t)(na + l15) * DH + half * 16;
        const __bf16* bpg = W1e + (size_t)(ng + l15) * DH + half * 16;
        for (int k0 = 0; k0 < DH; k0 += 32) {
            v16bf af = load_frag(ap  + k0, 8);   // A: K contiguous 16
            v16bf ba = load_frag(bpa + k0, 8);   // B: K contiguous 16
            v16bf bg = load_frag(bpg + k0, 8);
            acc_a = __builtin_amdgcn_wmma_f32_16x16x32_bf16(
                        false, af, false, ba, (short)0, acc_a, false, false);
            acc_g = __builtin_amdgcn_wmma_f32_16x16x32_bf16(
                        false, af, false, bg, (short)0, acc_g, false, false);
        }
        float bia = b1e[na + l15];
        float big = b1e[ng + l15];
#pragma unroll
        for (int r = 0; r < 8; ++r) {
            float a  = acc_a[r] + bia;
            float gt = acc_g[r] + big;
            float gv = a * (gt / (1.0f + __expf(-gt)));   // a * silu(gate)
            int m = r + half * 8;                         // C/D row layout
            gs[m * 2056 + na + l15] = (__bf16)gv;
        }
    }
    __syncthreads();

    const __bf16* W2e = W2t + (size_t)e * DH * INNER;
    const float*  b2e = b2 + (size_t)e * DH;

    // GEMM2: out = g @ W2 (16 x 2048 x 512); scale by router weight, atomic-add to y
    for (int nt = w; nt < DH / 16; nt += 8) {
        const int n0 = nt * 16;
        v8f acc = {};
        const __bf16* ap = gs + l15 * 2056 + half * 16;
        const __bf16* bp = W2e + (size_t)(n0 + l15) * INNER + half * 16;
        for (int k0 = 0; k0 < INNER; k0 += 32) {
            v16bf af = load_frag(ap + k0, 8);
            v16bf bf = load_frag(bp + k0, 8);
            acc = __builtin_amdgcn_wmma_f32_16x16x32_bf16(
                      false, af, false, bf, (short)0, acc, false, false);
        }
        float bb = b2e[n0 + l15];
#pragma unroll
        for (int r = 0; r < 8; ++r) {
            int m = r + half * 8;
            float v = (acc[r] + bb) * swt[m];
            unsafeAtomicAdd(&y[(size_t)stok[m] * DH + n0 + l15], v);
        }
    }
}

// ---------------------------------------------------------------------------------
extern "C" void kernel_launch(void* const* d_in, const int* in_sizes, int n_in,
                              void* d_out, int out_size, void* d_ws, size_t ws_size,
                              hipStream_t stream) {
    const float* x  = (const float*)d_in[0];
    const float* Wc = (const float*)d_in[1];
    const float* bc = (const float*)d_in[2];
    const float* W1 = (const float*)d_in[3];
    const float* b1 = (const float*)d_in[4];
    const float* W2 = (const float*)d_in[5];
    const float* b2 = (const float*)d_in[6];

    float* y      = (float*)d_out;
    float* logits = y + (size_t)T_TOK * DH;

    // Workspace layout
    char* ws = (char*)d_ws;
    const size_t w1t_bytes = (size_t)NEXP * F1 * DH * sizeof(short);     // 33.5 MB
    const size_t w2t_bytes = (size_t)NEXP * DH * INNER * sizeof(short);  // 16.8 MB
    __bf16* W1t = (__bf16*)ws;
    __bf16* W2t = (__bf16*)(ws + w1t_bytes);
    int*    cnt = (int*)(ws + w1t_bytes + w2t_bytes);
    int*    tokl = cnt + NEXP;
    float*  wgtl = (float*)(tokl + NEXP * T_TOK);

    // 1) zero y + expert counters
    moe_zero_kernel<<<(T_TOK * DH + 255) / 256, 256, 0, stream>>>(y, cnt, T_TOK * DH);

    // 2) weights fp32 -> bf16, transposed so K is contiguous per output column
    transpose_cvt_kernel<<<dim3(F1 / 32, DH / 32, NEXP), dim3(32, 8), 0, stream>>>(
        W1, W1t, DH, F1);
    transpose_cvt_kernel<<<dim3(DH / 32, INNER / 32, NEXP), dim3(32, 8), 0, stream>>>(
        W2, W2t, INNER, DH);

    // 3) router: logits + top-2 dispatch lists
    moe_router_kernel<<<T_TOK / 8, 256, 0, stream>>>(x, Wc, bc, logits, cnt, tokl, wgtl);

    // 4) grouped expert FFN (worst-case grid; blocks past the count early-out)
    moe_ffn_kernel<<<dim3(T_TOK / 16, NEXP), 256, 0, stream>>>(
        x, W1t, b1, W2t, b2, cnt, tokl, wgtl, y);
}
